// GATTransformer_84473416778133
// MI455X (gfx1250) — matrix operand order repaired and checked by reading, not compiled, not run on hardware
//
#include <hip/hip_runtime.h>
#include <hip/hip_bf16.h>
#include <math.h>

// ---------------- model constants ----------------
#define NN      4096      // nodes
#define NE      262144    // edges
#define F_IN    8
#define EMB     4
#define HEADS   4
#define C1      64        // GAT1 per-head channels (HC1 = 256)
#define HC1     256
#define C2      16        // GAT2 per-head channels (HC2 = 64 = D)
#define HC2     64
#define DM      64        // d_model
#define FF      32
#define GRAPHS  8
#define OUTD    4

typedef __attribute__((ext_vector_type(16))) _Float16 v16h;
typedef __attribute__((ext_vector_type(8)))  _Float16 v8h;
typedef __attribute__((ext_vector_type(8)))  float    v8f;

static inline int cdiv(int a, int b) { return (a + b - 1) / b; }

// ---------------- utility: fill u32 (also zeros f32) ----------------
__global__ void fill_u32_kernel(unsigned* __restrict__ p, unsigned v, int n) {
    int i = blockIdx.x * blockDim.x + threadIdx.x;
    if (i < n) p[i] = v;
}

// ---------------- convert + transpose weights: W[K x N] f32 -> Wt[N x K] f16 ----------------
__global__ void conv_transpose_kernel(const float* __restrict__ W, _Float16* __restrict__ Wt,
                                      int K, int N) {
    int i = blockIdx.x * blockDim.x + threadIdx.x;
    if (i >= K * N) return;
    int k = i / N, n = i % N;
    Wt[(size_t)n * K + k] = (_Float16)W[i];
}

// ---------------- fused feature_fc + GAT1 projection + attn dots ----------------
// one thread per (node, head): h1[n, hd*64+c] = (x@W_feat+b_feat) @ W1 ; s1/d1 = <h, a_s/a_d>
__global__ void gat1_h_sd_kernel(const float* __restrict__ x, const float* __restrict__ Wf,
                                 const float* __restrict__ bf, const float* __restrict__ W1,
                                 const float* __restrict__ as1, const float* __restrict__ ad1,
                                 float* __restrict__ h1, float* __restrict__ s1,
                                 float* __restrict__ d1) {
    int i = blockIdx.x * blockDim.x + threadIdx.x;
    if (i >= NN * HEADS) return;
    int n = i >> 2, hd = i & 3;
    float h0[EMB];
#pragma unroll
    for (int k = 0; k < EMB; ++k) {
        float acc = bf[k];
#pragma unroll
        for (int j = 0; j < F_IN; ++j) acc += x[n * F_IN + j] * Wf[j * EMB + k];
        h0[k] = acc;
    }
    float ss = 0.f, dd = 0.f;
    for (int c = 0; c < C1; ++c) {
        int col = hd * C1 + c;
        float hv = 0.f;
#pragma unroll
        for (int k = 0; k < EMB; ++k) hv += h0[k] * W1[k * HC1 + col];
        h1[(size_t)n * HC1 + col] = hv;
        ss += hv * as1[hd * C1 + c];
        dd += hv * ad1[hd * C1 + c];
    }
    s1[i] = ss;
    d1[i] = dd;
}

// ---------------- attn dots for layer 2 (h already computed by WMMA) ----------------
__global__ void sd_kernel(const float* __restrict__ h, const float* __restrict__ as,
                          const float* __restrict__ ad, float* __restrict__ s,
                          float* __restrict__ d, int H, int C) {
    int i = blockIdx.x * blockDim.x + threadIdx.x;
    if (i >= NN * H) return;
    int n = i / H, hd = i % H;
    const float* hp = h + (size_t)n * H * C + hd * C;
    float ss = 0.f, dd = 0.f;
    for (int c = 0; c < C; ++c) { ss += hp[c] * as[hd * C + c]; dd += hp[c] * ad[hd * C + c]; }
    s[i] = ss;
    d[i] = dd;
}

// monotonic float->uint key (order-preserving for atomicMax)
__device__ __forceinline__ unsigned f2key(float f) {
    unsigned b = __float_as_uint(f);
    return (b & 0x80000000u) ? ~b : (b | 0x80000000u);
}
__device__ __forceinline__ float key2f(unsigned k) {
    unsigned b = (k & 0x80000000u) ? (k ^ 0x80000000u) : ~k;
    return __uint_as_float(b);
}
#define KEY_NEG_INF 0x007FFFFFu   // f2key(-inf)

// ---------------- edge pass A: logits + segment max ----------------
__global__ void edge_logits_kernel(const int* __restrict__ src, const int* __restrict__ dst,
                                   const float* __restrict__ s, const float* __restrict__ d,
                                   float* __restrict__ e, unsigned* __restrict__ mx, int H) {
    int i = blockIdx.x * blockDim.x + threadIdx.x;
    if (i >= NE * H) return;
    int eid = i / H, hd = i % H;
    int sn = src[eid], dn = dst[eid];
    float lg = s[sn * H + hd] + d[dn * H + hd];
    float v = lg > 0.f ? lg : 0.2f * lg;   // leaky_relu(0.2), pre-softmax
    e[i] = v;
    atomicMax(&mx[dn * H + hd], f2key(v));
}

// ---------------- edge pass B: exp, denominator, weighted scatter of h[src] ----------------
__global__ void edge_accum_kernel(const int* __restrict__ src, const int* __restrict__ dst,
                                  const float* __restrict__ e, const unsigned* __restrict__ mx,
                                  const float* __restrict__ hfeat, float* __restrict__ den,
                                  float* __restrict__ num, int H, int C) {
    int i = blockIdx.x * blockDim.x + threadIdx.x;
    if (i >= NE * H) return;
    int eid = i / H, hd = i % H;
    int sn = src[eid], dn = dst[eid];
    float m = key2f(mx[dn * H + hd]);
    float ex = __expf(e[i] - m);
    atomicAdd(&den[dn * H + hd], ex);
    const float* hs = hfeat + (size_t)sn * H * C + hd * C;
    float* np = num + (size_t)dn * H * C + hd * C;
    for (int c = 0; c < C; ++c) atomicAdd(&np[c], ex * hs[c]);
}

// ---------------- GAT finalize: normalize, +bias, leaky_relu(0.01), emit f16 ----------------
__global__ void gat_finalize_kernel(const float* __restrict__ num, const float* __restrict__ den,
                                    const float* __restrict__ bias, _Float16* __restrict__ outh,
                                    int H, int C) {
    int HC = H * C;
    int i = blockIdx.x * blockDim.x + threadIdx.x;
    if (i >= NN * HC) return;
    int n = i / HC, j = i % HC;
    float dd = den[n * H + j / C];
    float v = num[i] / (dd + 1e-16f) + bias[j];
    v = v > 0.f ? v : 0.01f * v;
    outh[i] = (_Float16)v;
}

// ---------------- WMMA GEMM: C[MxN] = A[MxK] * Bt[NxK]^T (+bias)(+act) ----------------
// A row-major f16, Bt row-major f16 (i.e. B transposed). 1 wave = one 16x16 tile.
// act: 0=none, 1=relu, 2=leaky(0.01). Cf/Ch optional outputs.
__global__ void wmma_gemm_kernel(const _Float16* __restrict__ A, const _Float16* __restrict__ Bt,
                                 const float* __restrict__ bias, float* __restrict__ Cf,
                                 _Float16* __restrict__ Ch, int M, int N, int K, int act,
                                 int nTiles) {
    int tile = blockIdx.x * (blockDim.x >> 5) + (threadIdx.x >> 5);
    if (tile >= nTiles) return;                 // whole-wave exit: EXEC stays all-ones for WMMA
    int lane = threadIdx.x & 31;
    int tilesN = N >> 4;
    int mb = (tile / tilesN) << 4;
    int nb = (tile % tilesN) << 4;
    int r  = lane & 15;                         // row (A) / col (B) / col (D)
    int hi = lane >> 4;

    // ISA 7.12.2: A 16x32 f16 — lane half 'hi' holds K in {hi*8..hi*8+7, 16+hi*8..16+hi*8+7}
    const _Float16* arow = A  + (size_t)(mb + r) * K + (hi << 3);
    // B 32x16 f16 — same K order as A: lane half 'hi' holds K in {hi*8..hi*8+7, 16+hi*8..16+hi*8+7}
    const _Float16* brow = Bt + (size_t)(nb + r) * K + (hi << 3);

    v8f acc = {};
    for (int k0 = 0; k0 < K; k0 += 32) {
        v8h a0 = *(const v8h*)(arow + k0);
        v8h a1 = *(const v8h*)(arow + k0 + 16);
        v8h b0 = *(const v8h*)(brow + k0);
        v8h b1 = *(const v8h*)(brow + k0 + 16);
        v16h a, b;
#pragma unroll
        for (int t = 0; t < 8; ++t) {
            a[t] = a0[t]; a[t + 8] = a1[t];
            b[t] = b0[t]; b[t + 8] = b1[t];
        }
        acc = __builtin_amdgcn_wmma_f32_16x16x32_f16(false, a, false, b, (short)0, acc,
                                                     false, false);
    }
#pragma unroll
    for (int v = 0; v < 8; ++v) {
        int m = mb + v + (hi << 3);             // D layout: VGPR v -> M = v + 8*hi
        int n = nb + r;
        float val = acc[v];
        if (bias) val += bias[n];
        if (act == 1) val = fmaxf(val, 0.f);
        else if (act == 2) val = val > 0.f ? val : 0.01f * val;
        size_t idx = (size_t)m * N + n;
        if (Cf) Cf[idx] = val;
        if (Ch) Ch[idx] = (_Float16)val;
    }
}

// ---------------- mean-pool per graph ----------------
__global__ void pool_kernel(const float* __restrict__ t, const int* __restrict__ batch,
                            float* __restrict__ pooled, float* __restrict__ cnt) {
    int n = blockIdx.x * blockDim.x + threadIdx.x;
    if (n >= NN) return;
    int g = batch[n];
    atomicAdd(&cnt[g], 1.0f);
    const float* tp = t + (size_t)n * DM;
    for (int c = 0; c < DM; ++c) atomicAdd(&pooled[g * DM + c], tp[c]);
}

// ---------------- final fc: out[g,o] = (pooled[g]/max(cnt,1)) @ W_fc + b_fc ----------------
__global__ void fc_kernel(const float* __restrict__ pooled, const float* __restrict__ cnt,
                          const float* __restrict__ Wfc, const float* __restrict__ bfc,
                          float* __restrict__ out) {
    int i = threadIdx.x;
    if (i >= GRAPHS * OUTD) return;
    int g = i / OUTD, o = i % OUTD;
    float inv = 1.0f / fmaxf(cnt[g], 1.0f);
    float acc = bfc[o];
    for (int c = 0; c < DM; ++c) acc += pooled[g * DM + c] * inv * Wfc[c * OUTD + o];
    out[i] = acc;
}

extern "C" void kernel_launch(void* const* d_in, const int* in_sizes, int n_in,
                              void* d_out, int out_size, void* d_ws, size_t ws_size,
                              hipStream_t stream) {
    // inputs (setup_inputs order)
    const float* x      = (const float*)d_in[0];
    const int*   eidx   = (const int*)d_in[1];
    const int*   batch  = (const int*)d_in[2];
    const float* W_feat = (const float*)d_in[3];
    const float* b_feat = (const float*)d_in[4];
    const float* W1     = (const float*)d_in[5];
    const float* as1    = (const float*)d_in[6];
    const float* ad1    = (const float*)d_in[7];
    const float* b1     = (const float*)d_in[8];
    const float* W2     = (const float*)d_in[9];
    const float* as2    = (const float*)d_in[10];
    const float* ad2    = (const float*)d_in[11];
    const float* b2     = (const float*)d_in[12];
    // d_in[13..16]: enc_* — dead code in the reference (output never reads them)
    const float* dec_W1 = (const float*)d_in[17];
    const float* dec_b1 = (const float*)d_in[18];
    const float* dec_W2 = (const float*)d_in[19];
    const float* dec_b2 = (const float*)d_in[20];
    const float* W_fc   = (const float*)d_in[21];
    const float* b_fc   = (const float*)d_in[22];
    const int* src = eidx;            // edge_index[0]
    const int* dst = eidx + NE;       // edge_index[1]
    float* out = (float*)d_out;

    // workspace carve-out (256B aligned)
    char* base = (char*)d_ws;
    size_t off = 0;
    auto alloc = [&](size_t bytes) -> void* {
        void* p = base + off;
        off = (off + bytes + 255) & ~(size_t)255;
        return p;
    };
    float*     h1     = (float*)alloc((size_t)NN * HC1 * 4);   // also reused as num2
    float*     s1     = (float*)alloc((size_t)NN * HEADS * 4);
    float*     d1     = (float*)alloc((size_t)NN * HEADS * 4);
    float*     s2     = (float*)alloc((size_t)NN * HEADS * 4);
    float*     d2     = (float*)alloc((size_t)NN * HEADS * 4);
    float*     ebuf   = (float*)alloc((size_t)NE * HEADS * 4); // reused by both layers
    unsigned*  mx     = (unsigned*)alloc((size_t)NN * HEADS * 4);
    float*     den    = (float*)alloc((size_t)NN * HEADS * 4);
    float*     num1   = (float*)alloc((size_t)NN * HC1 * 4);   // also reused as t2
    _Float16*  a1h    = (_Float16*)alloc((size_t)NN * HC1 * 2);
    float*     h2     = (float*)alloc((size_t)NN * HC2 * 4);
    _Float16*  g2h    = (_Float16*)alloc((size_t)NN * HC2 * 2);
    _Float16*  t1h    = (_Float16*)alloc((size_t)NN * FF * 2);
    _Float16*  Wt2h   = (_Float16*)alloc((size_t)DM * HC1 * 2); // [64 x 256]
    _Float16*  d1th   = (_Float16*)alloc((size_t)FF * DM * 2);  // [32 x 64]
    _Float16*  d2th   = (_Float16*)alloc((size_t)DM * FF * 2);  // [64 x 32]
    float*     pooled = (float*)alloc((size_t)GRAPHS * DM * 4);
    float*     cnt    = (float*)alloc((size_t)GRAPHS * 4);
    float*     num2   = h1;      // layer-2 numerator aliases h1 (h1 dead by then)
    float*     t2     = num1;    // decoder output aliases num1 (dead by then)

    const int B = 256;
    const int NH  = NN * HEADS;
    const int EH  = NE * HEADS;

    // ---- prep: f16-transposed weights for WMMA ----
    conv_transpose_kernel<<<cdiv(HC1 * DM, B), B, 0, stream>>>(W2, Wt2h, HC1, DM);
    conv_transpose_kernel<<<cdiv(DM * FF, B), B, 0, stream>>>(dec_W1, d1th, DM, FF);
    conv_transpose_kernel<<<cdiv(FF * DM, B), B, 0, stream>>>(dec_W2, d2th, FF, DM);

    // ---- init accumulators (must happen every call) ----
    fill_u32_kernel<<<cdiv(NH, B), B, 0, stream>>>(mx, KEY_NEG_INF, NH);
    fill_u32_kernel<<<cdiv(NH, B), B, 0, stream>>>((unsigned*)den, 0u, NH);
    fill_u32_kernel<<<cdiv(NN * HC1, B), B, 0, stream>>>((unsigned*)num1, 0u, NN * HC1);
    fill_u32_kernel<<<1, B, 0, stream>>>((unsigned*)pooled, 0u, GRAPHS * DM);
    fill_u32_kernel<<<1, 32, 0, stream>>>((unsigned*)cnt, 0u, GRAPHS);

    // ---- layer 1: feature_fc + projection + attention ----
    gat1_h_sd_kernel<<<cdiv(NH, B), B, 0, stream>>>(x, W_feat, b_feat, W1, as1, ad1, h1, s1, d1);
    edge_logits_kernel<<<cdiv(EH, B), B, 0, stream>>>(src, dst, s1, d1, ebuf, mx, HEADS);
    edge_accum_kernel<<<cdiv(EH, B), B, 0, stream>>>(src, dst, ebuf, mx, h1, den, num1, HEADS, C1);
    gat_finalize_kernel<<<cdiv(NN * HC1, B), B, 0, stream>>>(num1, den, b1, a1h, HEADS, C1);

    // ---- h2 = a1 @ W2 : 4096x256 @ 256x64 (WMMA f16->f32) ----
    {
        int tiles = (NN / 16) * (HC2 / 16);   // 1024
        wmma_gemm_kernel<<<cdiv(tiles, 4), 128, 0, stream>>>(a1h, Wt2h, nullptr, h2, nullptr,
                                                             NN, HC2, HC1, 0, tiles);
    }

    // ---- layer 2 attention ----
    sd_kernel<<<cdiv(NH, B), B, 0, stream>>>(h2, as2, ad2, s2, d2, HEADS, C2);
    fill_u32_kernel<<<cdiv(NH, B), B, 0, stream>>>(mx, KEY_NEG_INF, NH);
    fill_u32_kernel<<<cdiv(NH, B), B, 0, stream>>>((unsigned*)den, 0u, NH);
    fill_u32_kernel<<<cdiv(NN * HC2, B), B, 0, stream>>>((unsigned*)num2, 0u, NN * HC2);
    edge_logits_kernel<<<cdiv(EH, B), B, 0, stream>>>(src, dst, s2, d2, ebuf, mx, HEADS);
    edge_accum_kernel<<<cdiv(EH, B), B, 0, stream>>>(src, dst, ebuf, mx, h2, den, num2, HEADS, C2);
    gat_finalize_kernel<<<cdiv(NN * HC2, B), B, 0, stream>>>(num2, den, b2, g2h, HEADS, C2);

    // ---- decoder FFN (encoder 'memory' is dead code — skipped) ----
    {
        int tiles = (NN / 16) * (FF / 16);    // 512
        wmma_gemm_kernel<<<cdiv(tiles, 4), 128, 0, stream>>>(g2h, d1th, dec_b1, nullptr, t1h,
                                                             NN, FF, DM, 1, tiles);
    }
    {
        int tiles = (NN / 16) * (DM / 16);    // 1024
        wmma_gemm_kernel<<<cdiv(tiles, 4), 128, 0, stream>>>(t1h, d2th, dec_b2, t2, nullptr,
                                                             NN, DM, FF, 1, tiles);
    }

    // ---- mean pool + final fc ----
    pool_kernel<<<cdiv(NN, B), B, 0, stream>>>(t2, batch, pooled, cnt);
    fc_kernel<<<1, 32, 0, stream>>>(pooled, cnt, W_fc, b_fc, out);
}
